// DGCNN_56478819942794
// MI455X (gfx1250) — matrix operand order repaired and checked by reading, not compiled, not run on hardware
//
#include <hip/hip_runtime.h>
#include <hip/hip_bf16.h>

// DGCNN forward for MI455X (gfx1250, wave32, WMMA).
// All GEMMs (pairwise-distance Gram matrix, edge-conv, fusion conv) run through
// v_wmma_f32_16x16x32_f16 with f16 operands / f32 accumulation.
// Layouts make every WMMA fragment two contiguous global_load_b128 per lane:
//   A (16x32): row-major weights Wh[o][C2pad]
//   B (32x16): K-major (transposed) activations EhT[col][C2pad]
// Each wave register-blocks 1x4 tiles (16x64 output strip): the A fragment is
// loaded once per K-step and reused for 4 WMMAs, raising wmma density per load
// (2.5 b128 loads / wmma) and quartering the BN-stat atomics.
// BN is two-pass (in-wave shuffle reduction + f32 atomics, then recompute +
// normalize). Max-over-k uses an exact order-independent float atomic max.
// Per-batch reuse of the 16MB dist buffer and <=20MB edge buffer keeps the
// working set inside the 192MB L2. Workspace requirement: ~111 MB.

#define BATCH 8
#define NPTS  2048
#define KNNK  20
#define NKCOL (NPTS * KNNK)   // 40960
#define COLS5 (BATCH * NPTS)  // 16384
#define EPSV  1e-5f
#define NEGS  0.2f

typedef __attribute__((ext_vector_type(16))) _Float16 v16h;
typedef __attribute__((ext_vector_type(8)))  _Float16 v8h;   // 16 bytes
typedef __attribute__((ext_vector_type(8)))  float    v8f;

union AFrag { v16h v; v8h q[2]; _Float16 h[16]; };
union CFrag { v8f  v; float f[8]; };

__device__ __forceinline__ void atomicMaxF32(float* addr, float v) {
    // Exact, order-independent float max. Works with -inf init.
    if (v >= 0.0f) atomicMax((int*)addr, __float_as_int(v));
    else           atomicMin((unsigned int*)addr, __float_as_uint(v));
}

__device__ __forceinline__ float lrelu(float y) {
    return y >= 0.0f ? y : NEGS * y;
}

// Load A fragment (16x32 f16, M x K): lane holds row (lane&15),
// K-halves {8g+0..7, 8g+16..23} relative to kk. src row-major, stride ld.
__device__ __forceinline__ void load_frag_a(const _Float16* __restrict__ src,
                                            int row, int ld, int kk, int lane,
                                            AFrag& a) {
    int g = lane >> 4, l15 = lane & 15;
    const _Float16* base = src + (size_t)(row + l15) * ld + kk + 8 * g;
    a.q[0] = *(const v8h*)(base);
    a.q[1] = *(const v8h*)(base + 16);
}

// Load B fragment (32x16 f16, K x N): lane holds column (lane&15),
// K-halves {8g+0..7, 8g+16..23} relative to kk. src is K-major per column, stride ld.
__device__ __forceinline__ void load_frag_b(const _Float16* __restrict__ src,
                                            int col, int ld, int kk, int lane,
                                            AFrag& b) {
    int g = lane >> 4, l15 = lane & 15;
    const _Float16* base = src + (size_t)(col + l15) * ld + kk + 8 * g;
    b.q[0] = *(const v8h*)(base);
    b.q[1] = *(const v8h*)(base + 16);
}

// 16x64 strip: D[j] (j=0..3) = A(16xK) x B[j](Kx16); A fragment reused 4x.
__device__ __forceinline__ void wmma_strip_1x4(const _Float16* __restrict__ Amat,
                                               const _Float16* __restrict__ Bmat,
                                               int Kpad, int mrow0, int ncol0,
                                               int lane, CFrag acc[4]) {
    for (int kk = 0; kk < Kpad; kk += 32) {
        AFrag a;
        load_frag_a(Amat, mrow0, Kpad, kk, lane, a);
#pragma unroll
        for (int j = 0; j < 4; ++j) {
            AFrag bf;
            load_frag_b(Bmat, ncol0 + 16 * j, Kpad, kk, lane, bf);
            acc[j].v = __builtin_amdgcn_wmma_f32_16x16x32_f16(
                false, a.v, false, bf.v, (short)0, acc[j].v, false, false);
        }
    }
}

// ---------------------------------------------------------------- utilities

__global__ void k_zero(float* __restrict__ p, int n) {
    int t = blockIdx.x * blockDim.x + threadIdx.x;
    if (t < n) p[t] = 0.0f;
}

__global__ void k_fill_neginf(float* __restrict__ p, int n) {
    int t = blockIdx.x * blockDim.x + threadIdx.x;
    if (t < n) p[t] = -__builtin_inff();
}

// Features [B][C][N] f32 -> point-major f16 xhT[b][n][Cpad] (zero pad channels).
__global__ void k_pad_f16(const float* __restrict__ x, _Float16* __restrict__ xh,
                          int C, int Cpad) {
    int t = blockIdx.x * blockDim.x + threadIdx.x;
    int total = BATCH * NPTS * Cpad;
    if (t >= total) return;
    int c = t % Cpad;
    int n = (t / Cpad) % NPTS;
    int b = t / (Cpad * NPTS);
    float v = (c < C) ? x[(b * C + c) * NPTS + n] : 0.0f;
    xh[t] = (_Float16)v;
}

// xx[b][n] = sum_c x[b][c][n]^2
__global__ void k_sumsq(const float* __restrict__ x, float* __restrict__ xx, int C) {
    int t = blockIdx.x * blockDim.x + threadIdx.x;
    if (t >= BATCH * NPTS) return;
    int n = t % NPTS, b = t / NPTS;
    float s = 0.0f;
    for (int c = 0; c < C; ++c) {
        float v = x[(b * C + c) * NPTS + n];
        s = fmaf(v, v, s);
    }
    xx[t] = s;
}

// W [O][C2] f32 -> Wh [O][C2pad] f16 (zero pad)
__global__ void k_cvt_w(const float* __restrict__ W, _Float16* __restrict__ Wh,
                        int O, int C2, int C2pad) {
    int t = blockIdx.x * blockDim.x + threadIdx.x;
    if (t >= O * C2pad) return;
    int c = t % C2pad, o = t / C2pad;
    Wh[t] = (_Float16)((c < C2) ? W[o * C2 + c] : 0.0f);
}

// ------------------------------------------------- pairwise distance (WMMA)

// neg_dist 16x64 strip = 2 * (X X^T) - xx[row] - xx[col]. xh: [N][Cpad] f16.
__global__ void k_dist_wmma(const _Float16* __restrict__ xh,
                            const float* __restrict__ xx,
                            float* __restrict__ dist, int Cpad) {
    const int mtiles = NPTS / 16;   // 128
    const int ntile4 = NPTS / 64;   // 32
    int wid  = blockIdx.x * (blockDim.x >> 5) + (threadIdx.x >> 5);
    if (wid >= mtiles * ntile4) return;                // wave-uniform
    int lane = threadIdx.x & 31;
    int g = lane >> 4, l15 = lane & 15;
    int mt = wid / ntile4, nt = wid % ntile4;
    int row0 = mt * 16, col0 = nt * 64;

    CFrag acc[4];
#pragma unroll
    for (int j = 0; j < 4; ++j)
#pragma unroll
        for (int r = 0; r < 8; ++r) acc[j].f[r] = 0.0f;

    wmma_strip_1x4(xh, xh, Cpad, row0, col0, lane, acc);

#pragma unroll
    for (int j = 0; j < 4; ++j) {
        int col = col0 + 16 * j + l15;
        float xxc = xx[col];
#pragma unroll
        for (int r = 0; r < 8; ++r) {
            int row = row0 + 8 * g + r;
            dist[row * NPTS + col] = 2.0f * acc[j].f[r] - xx[row] - xxc;
        }
    }
}

// Top-KNNK per row via 20 wave-wide argmax passes (wave32 shuffles).
__global__ void k_topk(const float* __restrict__ dist, int* __restrict__ idx) {
    __shared__ float row[NPTS];
    int n = blockIdx.x;
    int lane = threadIdx.x;
    const float* dr = dist + (size_t)n * NPTS;
    for (int j = lane; j < NPTS; j += 32) row[j] = dr[j];
    __syncthreads();
    for (int sel = 0; sel < KNNK; ++sel) {
        float bv = -__builtin_inff();
        int   bi = 0;
        for (int j = lane; j < NPTS; j += 32) {
            float v = row[j];
            if (v > bv) { bv = v; bi = j; }
        }
#pragma unroll
        for (int m = 16; m >= 1; m >>= 1) {
            float ov = __shfl_xor(bv, m, 32);
            int   oi = __shfl_xor(bi, m, 32);
            if (ov > bv || (ov == bv && oi < bi)) { bv = ov; bi = oi; }
        }
        if (lane == 0) { idx[n * KNNK + sel] = bi; row[bi] = -__builtin_inff(); }
        __syncthreads();
    }
}

// -------------------------------------------------------- edge features

// EhT [N*K][C2pad] f16 for one batch (K-major per column -> contiguous B frags):
// c<C -> x[idx]-x[n]; c<2C -> x[n]; else 0. Writes fully coalesced along c2.
__global__ void k_build_edge(const float* __restrict__ xin, const int* __restrict__ idx,
                             _Float16* __restrict__ Eh, int C, int C2pad) {
    int t = blockIdx.x * blockDim.x + threadIdx.x;
    if (t >= NKCOL * C2pad) return;
    int c2  = t % C2pad;
    int col = t / C2pad;
    int n = col / KNNK, k = col % KNNK;
    float v;
    if (c2 < C) {
        int j = idx[n * KNNK + k];
        v = xin[c2 * NPTS + j] - xin[c2 * NPTS + n];
    } else if (c2 < 2 * C) {
        v = xin[(c2 - C) * NPTS + n];
    } else {
        v = 0.0f;
    }
    Eh[t] = (_Float16)v;
}

// ------------------------------------------------------------ GEMM kernels

// Pass 1: per-channel sum / sum-of-squares of y = W @ E over a 16x64 strip.
// All 4 tiles share the same 16 rows (channels) -> one fused reduction.
__global__ void k_gemm_stats(const _Float16* __restrict__ Wh, const _Float16* __restrict__ Eh,
                             float* __restrict__ sum, float* __restrict__ sumsq,
                             int C2pad, int mtiles, int ntile4) {
    int wid = blockIdx.x * (blockDim.x >> 5) + (threadIdx.x >> 5);
    if (wid >= mtiles * ntile4) return;
    int lane = threadIdx.x & 31;
    int mt = wid / ntile4, nt = wid % ntile4;

    CFrag acc[4];
#pragma unroll
    for (int j = 0; j < 4; ++j)
#pragma unroll
        for (int r = 0; r < 8; ++r) acc[j].f[r] = 0.0f;
    wmma_strip_1x4(Wh, Eh, C2pad, mt * 16, nt * 64, lane, acc);

    float s1[8], s2[8];
#pragma unroll
    for (int r = 0; r < 8; ++r) { s1[r] = 0.0f; s2[r] = 0.0f; }
#pragma unroll
    for (int j = 0; j < 4; ++j)
#pragma unroll
        for (int r = 0; r < 8; ++r) {
            float y = acc[j].f[r];
            s1[r] += y;
            s2[r] = fmaf(y, y, s2[r]);
        }
#pragma unroll
    for (int m = 1; m < 16; m <<= 1) {
#pragma unroll
        for (int r = 0; r < 8; ++r) {
            s1[r] += __shfl_xor(s1[r], m, 32);
            s2[r] += __shfl_xor(s2[r], m, 32);
        }
    }
    if ((lane & 15) == 0) {
        int rbase = mt * 16 + 8 * (lane >> 4);
#pragma unroll
        for (int r = 0; r < 8; ++r) {
            atomicAdd(&sum[rbase + r],   s1[r]);
            atomicAdd(&sumsq[rbase + r], s2[r]);
        }
    }
}

__global__ void k_stats_fin(const float* __restrict__ sum, const float* __restrict__ sumsq,
                            float* __restrict__ mean, float* __restrict__ rstd,
                            int O, float invcnt) {
    int o = blockIdx.x * blockDim.x + threadIdx.x;
    if (o >= O) return;
    float m = sum[o] * invcnt;
    float v = sumsq[o] * invcnt - m * m;
    mean[o] = m;
    rstd[o] = rsqrtf(v + EPSV);
}

// Pass 2 (edge conv): recompute strip, BN + lrelu, max-reduce over k into xout[O][N].
__global__ void k_gemm_max(const _Float16* __restrict__ Wh, const _Float16* __restrict__ Eh,
                           const float* __restrict__ mean, const float* __restrict__ rstd,
                           const float* __restrict__ gma, const float* __restrict__ bta,
                           float* __restrict__ xout, int C2pad, int mtiles, int ntile4) {
    int wid = blockIdx.x * (blockDim.x >> 5) + (threadIdx.x >> 5);
    if (wid >= mtiles * ntile4) return;
    int lane = threadIdx.x & 31;
    int g = lane >> 4, l15 = lane & 15;
    int mt = wid / ntile4, nt = wid % ntile4;

    CFrag acc[4];
#pragma unroll
    for (int j = 0; j < 4; ++j)
#pragma unroll
        for (int r = 0; r < 8; ++r) acc[j].f[r] = 0.0f;
    wmma_strip_1x4(Wh, Eh, C2pad, mt * 16, nt * 64, lane, acc);

    float mn[8], rs[8], gm[8], bt[8];
#pragma unroll
    for (int r = 0; r < 8; ++r) {
        int o = mt * 16 + 8 * g + r;
        mn[r] = mean[o]; rs[r] = rstd[o]; gm[r] = gma[o]; bt[r] = bta[o];
    }
#pragma unroll
    for (int j = 0; j < 4; ++j) {
        int col = nt * 64 + 16 * j + l15;
        int n = col / KNNK;
#pragma unroll
        for (int r = 0; r < 8; ++r) {
            int o = mt * 16 + 8 * g + r;
            float y = lrelu((acc[j].f[r] - mn[r]) * rs[r] * gm[r] + bt[r]);
            atomicMaxF32(&xout[o * NPTS + n], y);
        }
    }
}

// Pass 2 (fusion conv, layer 5): write normalized activations to x5[B][512][N].
__global__ void k_gemm_x5(const _Float16* __restrict__ Wh, const _Float16* __restrict__ Eh,
                          const float* __restrict__ mean, const float* __restrict__ rstd,
                          const float* __restrict__ gma, const float* __restrict__ bta,
                          float* __restrict__ x5, int C2pad, int mtiles, int ntile4) {
    int wid = blockIdx.x * (blockDim.x >> 5) + (threadIdx.x >> 5);
    if (wid >= mtiles * ntile4) return;
    int lane = threadIdx.x & 31;
    int g = lane >> 4, l15 = lane & 15;
    int mt = wid / ntile4, nt = wid % ntile4;

    CFrag acc[4];
#pragma unroll
    for (int j = 0; j < 4; ++j)
#pragma unroll
        for (int r = 0; r < 8; ++r) acc[j].f[r] = 0.0f;
    wmma_strip_1x4(Wh, Eh, C2pad, mt * 16, nt * 64, lane, acc);

    float mn[8], rs[8], gm[8], bt[8];
#pragma unroll
    for (int r = 0; r < 8; ++r) {
        int o = mt * 16 + 8 * g + r;
        mn[r] = mean[o]; rs[r] = rstd[o]; gm[r] = gma[o]; bt[r] = bta[o];
    }
#pragma unroll
    for (int j = 0; j < 4; ++j) {
        int col = nt * 64 + 16 * j + l15;
        int b = col >> 11, n = col & (NPTS - 1);
#pragma unroll
        for (int r = 0; r < 8; ++r) {
            int o = mt * 16 + 8 * g + r;
            float y = lrelu((acc[j].f[r] - mn[r]) * rs[r] * gm[r] + bt[r]);
            x5[((size_t)b * 512 + o) * NPTS + n] = y;
        }
    }
}

// Concatenate x1..x4 -> xcT [B*N][512] f16 (K-major for B fragments).
__global__ void k_concat5(const float* __restrict__ x1, const float* __restrict__ x2,
                          const float* __restrict__ x3, const float* __restrict__ x4,
                          _Float16* __restrict__ xc) {
    int t = blockIdx.x * blockDim.x + threadIdx.x;
    if (t >= COLS5 * 512) return;
    int c   = t % 512;
    int col = t / 512;
    int b = col >> 11, n = col & (NPTS - 1);
    float v;
    if      (c < 64)  v = x1[((size_t)b * 64  + c)        * NPTS + n];
    else if (c < 128) v = x2[((size_t)b * 64  + (c - 64)) * NPTS + n];
    else if (c < 256) v = x3[((size_t)b * 128 + (c - 128)) * NPTS + n];
    else              v = x4[((size_t)b * 256 + (c - 256)) * NPTS + n];
    xc[t] = (_Float16)v;
}

// Global max + mean pooling over N per (b, o).
__global__ void k_pool(const float* __restrict__ x5, float* __restrict__ feat) {
    __shared__ float smax[256];
    __shared__ float ssum[256];
    int bo = blockIdx.x;
    int b = bo / 512, o = bo % 512;
    const float* p = x5 + ((size_t)b * 512 + o) * NPTS;
    int tid = threadIdx.x;
    float mx = -__builtin_inff(), sm = 0.0f;
    for (int j = tid; j < NPTS; j += 256) {
        float v = p[j];
        mx = fmaxf(mx, v);
        sm += v;
    }
    smax[tid] = mx; ssum[tid] = sm;
    __syncthreads();
    for (int s = 128; s > 0; s >>= 1) {
        if (tid < s) {
            smax[tid] = fmaxf(smax[tid], smax[tid + s]);
            ssum[tid] += ssum[tid + s];
        }
        __syncthreads();
    }
    if (tid == 0) {
        feat[b * 1024 + o]       = smax[0];
        feat[b * 1024 + 512 + o] = ssum[0] * (1.0f / NPTS);
    }
}

// pre[b][o] = feat[b] . Wl[o]   (8x512 dots of length 1024 — tiny)
__global__ void k_fc(const float* __restrict__ feat, const float* __restrict__ Wl,
                     float* __restrict__ pre) {
    int t = blockIdx.x * blockDim.x + threadIdx.x;
    if (t >= BATCH * 512) return;
    int o = t % 512, b = t / 512;
    float s = 0.0f;
    for (int c = 0; c < 1024; ++c)
        s = fmaf(feat[b * 1024 + c], Wl[o * 1024 + c], s);
    pre[t] = s;
}

// Final BN over batch axis + lrelu.
__global__ void k_final(const float* __restrict__ pre, const float* __restrict__ g6,
                        const float* __restrict__ b6, float* __restrict__ out) {
    int o = blockIdx.x * blockDim.x + threadIdx.x;
    if (o >= 512) return;
    float m = 0.0f;
    for (int b = 0; b < BATCH; ++b) m += pre[b * 512 + o];
    m *= (1.0f / BATCH);
    float v = 0.0f;
    for (int b = 0; b < BATCH; ++b) {
        float d = pre[b * 512 + o] - m;
        v = fmaf(d, d, v);
    }
    v *= (1.0f / BATCH);
    float rs = rsqrtf(v + EPSV);
    for (int b = 0; b < BATCH; ++b) {
        float y = (pre[b * 512 + o] - m) * rs * g6[o] + b6[o];
        out[b * 512 + o] = lrelu(y);
    }
}

// ---------------------------------------------------------------- driver

extern "C" void kernel_launch(void* const* d_in, const int* in_sizes, int n_in,
                              void* d_out, int out_size, void* d_ws, size_t ws_size,
                              hipStream_t stream) {
    (void)in_sizes; (void)n_in; (void)out_size; (void)ws_size;

    const float* x   = (const float*)d_in[0];
    const float* W1  = (const float*)d_in[1];
    const float* g1  = (const float*)d_in[2];
    const float* b1  = (const float*)d_in[3];
    const float* W2  = (const float*)d_in[4];
    const float* g2  = (const float*)d_in[5];
    const float* b2  = (const float*)d_in[6];
    const float* W3  = (const float*)d_in[7];
    const float* g3  = (const float*)d_in[8];
    const float* b3  = (const float*)d_in[9];
    const float* W4  = (const float*)d_in[10];
    const float* g4  = (const float*)d_in[11];
    const float* b4  = (const float*)d_in[12];
    const float* W5  = (const float*)d_in[13];
    const float* g5  = (const float*)d_in[14];
    const float* b5  = (const float*)d_in[15];
    const float* Wl  = (const float*)d_in[16];
    const float* g6  = (const float*)d_in[17];
    const float* b6  = (const float*)d_in[18];
    float* out = (float*)d_out;

    // ---- workspace carving (~111 MB, 256B aligned regions) ----
    char* ws = (char*)d_ws;
    size_t off = 0;
    auto carve = [&](size_t bytes) -> char* {
        char* p = ws + off;
        off += (bytes + 255) & ~(size_t)255;
        return p;
    };
    float*    xx   = (float*)   carve((size_t)BATCH * NPTS * 4);
    int*      idx  = (int*)     carve((size_t)BATCH * NPTS * KNNK * 4);
    float*    dist = (float*)   carve((size_t)NPTS * NPTS * 4);           // per-batch reuse
    _Float16* xh   = (_Float16*)carve((size_t)BATCH * NPTS * 128 * 2);    // [b][n][Cpad]
    _Float16* Wh   = (_Float16*)carve((size_t)512 * 512 * 2);
    _Float16* Eh   = (_Float16*)carve((size_t)NKCOL * 256 * 2);           // also xcT (16MB)
    float*    ssum = (float*)   carve(512 * 4);
    float*    ssq  = (float*)   carve(512 * 4);
    float*    mean = (float*)   carve(512 * 4);
    float*    rstd = (float*)   carve(512 * 4);
    float*    x1   = (float*)   carve((size_t)BATCH * 64  * NPTS * 4);
    float*    x2   = (float*)   carve((size_t)BATCH * 64  * NPTS * 4);
    float*    x3   = (float*)   carve((size_t)BATCH * 128 * NPTS * 4);
    float*    x4   = (float*)   carve((size_t)BATCH * 256 * NPTS * 4);
    float*    x5   = (float*)   carve((size_t)BATCH * 512 * NPTS * 4);
    float*    feat = (float*)   carve((size_t)BATCH * 1024 * 4);
    float*    pre  = (float*)   carve((size_t)BATCH * 512 * 4);

    auto run_edge = [&](const float* xin, int C, int Cpad, int O,
                        const float* W, const float* gmy, const float* bty,
                        float* xout) {
        const int C2 = 2 * C;
        const int C2pad = ((C2 + 31) / 32) * 32;

        // kNN: features -> point-major f16, squared norms, dist GEMM + top-k.
        {
            int tot = BATCH * NPTS * Cpad;
            k_pad_f16<<<(tot + 255) / 256, 256, 0, stream>>>(xin, xh, C, Cpad);
        }
        k_sumsq<<<(BATCH * NPTS + 255) / 256, 256, 0, stream>>>(xin, xx, C);
        const int dwaves = (NPTS / 16) * (NPTS / 64);   // 128 * 32
        for (int b = 0; b < BATCH; ++b) {
            k_dist_wmma<<<(dwaves + 3) / 4, 128, 0, stream>>>(
                xh + (size_t)b * NPTS * Cpad, xx + b * NPTS, dist, Cpad);
            k_topk<<<NPTS, 32, 0, stream>>>(dist, idx + (size_t)b * NPTS * KNNK);
        }

        // Weight to f16 (padded).
        {
            int tot = O * C2pad;
            k_cvt_w<<<(tot + 255) / 256, 256, 0, stream>>>(W, Wh, O, C2, C2pad);
        }

        // Pass 1: BN stats.
        k_zero<<<2, 256, 0, stream>>>(ssum, 512);
        k_zero<<<2, 256, 0, stream>>>(ssq, 512);
        const int mtiles = O / 16, ntile4 = NKCOL / 64;   // 640
        for (int b = 0; b < BATCH; ++b) {
            int tot = NKCOL * C2pad;
            k_build_edge<<<(tot + 255) / 256, 256, 0, stream>>>(
                xin + (size_t)b * C * NPTS, idx + (size_t)b * NPTS * KNNK, Eh, C, C2pad);
            k_gemm_stats<<<(mtiles * ntile4 + 3) / 4, 128, 0, stream>>>(
                Wh, Eh, ssum, ssq, C2pad, mtiles, ntile4);
        }
        k_stats_fin<<<(O + 255) / 256, 256, 0, stream>>>(
            ssum, ssq, mean, rstd, O, 1.0f / ((float)BATCH * (float)NKCOL));

        // Pass 2: recompute, BN + lrelu, max over k.
        {
            int tot = BATCH * O * NPTS;
            k_fill_neginf<<<(tot + 255) / 256, 256, 0, stream>>>(xout, tot);
        }
        for (int b = 0; b < BATCH; ++b) {
            int tot = NKCOL * C2pad;
            k_build_edge<<<(tot + 255) / 256, 256, 0, stream>>>(
                xin + (size_t)b * C * NPTS, idx + (size_t)b * NPTS * KNNK, Eh, C, C2pad);
            k_gemm_max<<<(mtiles * ntile4 + 3) / 4, 128, 0, stream>>>(
                Wh, Eh, mean, rstd, gmy, bty, xout + (size_t)b * O * NPTS,
                C2pad, mtiles, ntile4);
        }
    };

    run_edge(x,   3,  32,  64, W1, g1, b1, x1);
    run_edge(x1, 64,  64,  64, W2, g2, b2, x2);
    run_edge(x2, 64,  64, 128, W3, g3, b3, x3);
    run_edge(x3, 128, 128, 256, W4, g4, b4, x4);

    // ---- layer 5: fusion conv over concat(x1..x4) ----
    {
        int tot = COLS5 * 512;
        k_concat5<<<(tot + 255) / 256, 256, 0, stream>>>(x1, x2, x3, x4, Eh);
    }
    {
        int tot = 512 * 512;
        k_cvt_w<<<(tot + 255) / 256, 256, 0, stream>>>(W5, Wh, 512, 512, 512);
    }
    k_zero<<<2, 256, 0, stream>>>(ssum, 512);
    k_zero<<<2, 256, 0, stream>>>(ssq, 512);
    {
        const int mt5 = 512 / 16, nt5 = COLS5 / 64;   // 32, 256
        k_gemm_stats<<<(mt5 * nt5 + 3) / 4, 128, 0, stream>>>(
            Wh, Eh, ssum, ssq, 512, mt5, nt5);
        k_stats_fin<<<2, 256, 0, stream>>>(ssum, ssq, mean, rstd, 512,
                                           1.0f / (float)COLS5);
        k_gemm_x5<<<(mt5 * nt5 + 3) / 4, 128, 0, stream>>>(
            Wh, Eh, mean, rstd, g5, b5, x5, 512, mt5, nt5);
    }

    // ---- pooling + final FC + BN ----
    k_pool<<<BATCH * 512, 256, 0, stream>>>(x5, feat);
    k_fc<<<(BATCH * 512 + 255) / 256, 256, 0, stream>>>(feat, Wl, pre);
    k_final<<<2, 256, 0, stream>>>(pre, g6, b6, out);
}
